// msa_3d_17386027614668
// MI455X (gfx1250) — matrix operand order repaired and checked by reading, not compiled, not run on hardware
//
#include <hip/hip_runtime.h>
#include <math.h>

typedef __attribute__((ext_vector_type(16))) _Float16 v16h;
typedef __attribute__((ext_vector_type(8)))  _Float16 v8h;
typedef __attribute__((ext_vector_type(8)))  float    v8f;

#define N_TOK 4096
#define C_IN  256
#define NHEAD 8
#define DHEAD 32

// scale = 1/sqrt(32) * log2(e): fold log2e into Q so softmax uses raw v_exp_f32
#define QSCALE 0.25499662792795535f

static __device__ __forceinline__ v16h make_a(v8h lo, v8h hi) {
  v16h a;
#pragma unroll
  for (int i = 0; i < 8; ++i) { a[i] = lo[i]; a[8 + i] = hi[i]; }
  return a;
}

static __device__ __forceinline__ v8f wmma_f16(v16h a, v16h b, v8f c) {
  return __builtin_amdgcn_wmma_f32_16x16x32_f16(false, a, false, b,
                                                (short)0, c, false, false);
}

// pack two f32 -> two f16 in one dword (v_cvt_pk_rtz_f16_f32)
static __device__ __forceinline__ int pk2h(float a, float b) {
  auto v = __builtin_amdgcn_cvt_pkrtz(a, b);
  int r;
  __builtin_memcpy(&r, &v, 4);
  return r;
}

// ---------------- conversion kernels ----------------

__global__ void cvt_x_f16(const float* __restrict__ x,
                          _Float16* __restrict__ xh, int n) {
  int i = blockIdx.x * blockDim.x + threadIdx.x;
  if (i < n) xh[i] = (_Float16)x[i];
}

// w: [C][F] row-major f32 -> wT: [F][C] f16
__global__ void cvt_w_transpose(const float* __restrict__ w,
                                _Float16* __restrict__ wT) {
  int c = blockIdx.x;      // 0..255
  int f = threadIdx.x;     // 0..255
  wT[f * C_IN + c] = (_Float16)w[c * C_IN + f];
}

// ---------------- QKV projection (WMMA) ----------------
// xh:[4096][256] f16 ; w*T:[256][256] f16 (transposed)
// qh:[NH][N][32] f16 (pre-scaled by QSCALE), kh:[NH][N][32] f16,
// vT:[NH][32][N] f16 (transposed: A-operand layout for O^T = V^T x P^T)
__global__ __launch_bounds__(32)
void qkv_proj(const _Float16* __restrict__ xh,
              const _Float16* __restrict__ wqT,
              const _Float16* __restrict__ wkT,
              const _Float16* __restrict__ wvT,
              const float* __restrict__ bq,
              const float* __restrict__ bk,
              const float* __restrict__ bv,
              _Float16* __restrict__ qh,
              _Float16* __restrict__ kh,
              _Float16* __restrict__ vT) {
  const int lane = threadIdx.x & 31;
  const int n  = lane & 15;
  const int hi = lane >> 4;
  const int m0 = blockIdx.x * 16;
  const int f0 = blockIdx.y * 16;

  v8f aq = {}, ak = {}, av = {};
  for (int kc = 0; kc < C_IN; kc += 32) {
    const _Float16* ap = xh + (m0 + n) * C_IN + kc;
    v16h A  = make_a(*(const v8h*)(ap + hi * 16),
                     *(const v8h*)(ap + hi * 16 + 8));
    v16h Bq = *(const v16h*)(wqT + (f0 + n) * C_IN + kc + hi * 16);
    v16h Bk = *(const v16h*)(wkT + (f0 + n) * C_IN + kc + hi * 16);
    v16h Bv = *(const v16h*)(wvT + (f0 + n) * C_IN + kc + hi * 16);
    aq = wmma_f16(A, Bq, aq);
    ak = wmma_f16(A, Bk, ak);
    av = wmma_f16(A, Bv, av);
  }

  const int h  = f0 >> 5;        // head
  const int d0 = f0 & 31;        // 0 or 16 within head
  const float bqv = bq[f0 + n];
  const float bkv = bk[f0 + n];
  const float bvv = bv[f0 + n];

#pragma unroll
  for (int j = 0; j < 8; ++j) {
    int row = m0 + hi * 8 + j;
    qh[(h * N_TOK + row) * DHEAD + d0 + n] = (_Float16)((aq[j] + bqv) * QSCALE);
    kh[(h * N_TOK + row) * DHEAD + d0 + n] = (_Float16)(ak[j] + bkv);
  }
  // transposed V store: for fixed lane, j indexes consecutive tokens -> packed
  v8h vp;
#pragma unroll
  for (int j = 0; j < 8; ++j) vp[j] = (_Float16)(av[j] + bvv);
  *(v8h*)(vT + (h * DHEAD + d0 + n) * N_TOK + m0 + hi * 8) = vp;
}

// ---------------- Flash attention, transposed form (WMMA, no LDS) ----------
// S^T = K_tile x Q^T ; softmax rows live in lane pairs (n, n+16);
// O^T = V^T x P^T with P^T built in-register via xor-16 lane exchange.
__global__ __launch_bounds__(32)
void attn_fwd(const _Float16* __restrict__ qh,
              const _Float16* __restrict__ kh,
              const _Float16* __restrict__ vT,
              _Float16* __restrict__ ao) {
  const int lane = threadIdx.x & 31;
  const int n  = lane & 15;
  const int hi = lane >> 4;
  const int h  = blockIdx.y;
  const int m0 = blockIdx.x * 16;

  // Q^T as B operand: lane = query m0+n, 16 contiguous dims at hi*16
  const v16h Bq =
      *(const v16h*)(qh + (size_t)(h * N_TOK + m0 + n) * DHEAD + hi * 16);

  const _Float16* khh = kh + (size_t)h * N_TOK * DHEAD;
  const _Float16* vp0 = vT + (size_t)(h * DHEAD + n) * N_TOK;        // dv = n
  const _Float16* vp1 = vT + (size_t)(h * DHEAD + 16 + n) * N_TOK;   // dv = 16+n

  v8f o0 = {}, o1 = {};
  float mr = -3.0e38f, lr = 0.0f;   // per-lane: stats of query m0+n

  for (int k0 = 0; k0 < N_TOK; k0 += 32) {
    // K tiles as A operands: lane = key row, contiguous dims
    const _Float16* kp0 = khh + (size_t)(k0 + n) * DHEAD;
    const _Float16* kp1 = khh + (size_t)(k0 + 16 + n) * DHEAD;
    v16h A0 = make_a(*(const v8h*)(kp0 + hi * 16),
                     *(const v8h*)(kp0 + hi * 16 + 8));
    v16h A1 = make_a(*(const v8h*)(kp1 + hi * 16),
                     *(const v8h*)(kp1 + hi * 16 + 8));
    v8f s0 = {}, s1 = {};
    s0 = wmma_f16(A0, Bq, s0);   // S^T keys [k0,k0+16)
    s1 = wmma_f16(A1, Bq, s1);   // S^T keys [k0+16,k0+32)

    // row (= this lane's query) max over 16 local elems + 1 lane exchange
    float t = fmaxf(s0[0], s1[0]);
#pragma unroll
    for (int j = 1; j < 8; ++j) t = fmaxf(t, fmaxf(s0[j], s1[j]));
    t = fmaxf(t, __shfl_xor(t, 16));
    float mnew  = fmaxf(mr, t);
    float alpha = __builtin_amdgcn_exp2f(mr - mnew);

    float p0[8], p1[8];
    float rs = 0.0f;
#pragma unroll
    for (int j = 0; j < 8; ++j) {
      p0[j] = __builtin_amdgcn_exp2f(s0[j] - mnew);
      p1[j] = __builtin_amdgcn_exp2f(s1[j] - mnew);
      rs += p0[j] + p1[j];
    }
    rs += __shfl_xor(rs, 16);
    lr = lr * alpha + rs;
    mr = mnew;
#pragma unroll
    for (int j = 0; j < 8; ++j) { o0[j] *= alpha; o1[j] *= alpha; }

    // Build P^T B-operand: pack f16 pairs, exchange across hi halves (xor 16)
    int pk0[4], pk1[4];
#pragma unroll
    for (int i = 0; i < 4; ++i) {
      pk0[i] = pk2h(p0[2 * i], p0[2 * i + 1]);
      pk1[i] = pk2h(p1[2 * i], p1[2 * i + 1]);
    }
    union { v16h h; int d[8]; } Bp;
#pragma unroll
    for (int i = 0; i < 4; ++i) {
      int x0 = __shfl_xor(pk0[i], 16);
      int x1 = __shfl_xor(pk1[i], 16);
      Bp.d[i]     = hi ? x1     : pk0[i];  // keys hi*16 + [0,8)
      Bp.d[4 + i] = hi ? pk1[i] : x0;      // keys hi*16 + [8,16)
    }

    // V^T tiles as A operands: lane = dv row, contiguous keys
    v16h Av0 = make_a(*(const v8h*)(vp0 + k0 + hi * 16),
                      *(const v8h*)(vp0 + k0 + hi * 16 + 8));
    v16h Av1 = make_a(*(const v8h*)(vp1 + k0 + hi * 16),
                      *(const v8h*)(vp1 + k0 + hi * 16 + 8));
    o0 = wmma_f16(Av0, Bp.h, o0);   // O^T dv [0,16)
    o1 = wmma_f16(Av1, Bp.h, o1);   // O^T dv [16,32)
  }

  // O^T element j: (dv = hi*8+j [+16 for o1], query = n) -> packed stores
  const float inv = 1.0f / lr;
  v8h r0, r1;
#pragma unroll
  for (int j = 0; j < 8; ++j) {
    r0[j] = (_Float16)(o0[j] * inv);
    r1[j] = (_Float16)(o1[j] * inv);
  }
  _Float16* op = ao + (size_t)(m0 + n) * 256 + h * DHEAD;
  *(v8h*)(op + hi * 8)      = r0;
  *(v8h*)(op + 16 + hi * 8) = r1;
}

// ---------------- Output projection (WMMA) ----------------
__global__ __launch_bounds__(32)
void out_proj(const _Float16* __restrict__ ah,
              const _Float16* __restrict__ woT,
              const float* __restrict__ bo,
              float* __restrict__ out) {
  const int lane = threadIdx.x & 31;
  const int n  = lane & 15;
  const int hi = lane >> 4;
  const int m0 = blockIdx.x * 16;
  const int f0 = blockIdx.y * 16;

  v8f acc = {};
  for (int kc = 0; kc < C_IN; kc += 32) {
    const _Float16* ap = ah + (m0 + n) * C_IN + kc;
    v16h A = make_a(*(const v8h*)(ap + hi * 16),
                    *(const v8h*)(ap + hi * 16 + 8));
    v16h B = *(const v16h*)(woT + (f0 + n) * C_IN + kc + hi * 16);
    acc = wmma_f16(A, B, acc);
  }
  const float bv = bo[f0 + n];
#pragma unroll
  for (int j = 0; j < 8; ++j)
    out[(m0 + hi * 8 + j) * 256 + f0 + n] = acc[j] + bv;
}

// ---------------- launcher ----------------
extern "C" void kernel_launch(void* const* d_in, const int* in_sizes, int n_in,
                              void* d_out, int out_size, void* d_ws, size_t ws_size,
                              hipStream_t stream) {
  const float* x  = (const float*)d_in[0];
  const float* wq = (const float*)d_in[1];
  const float* bq = (const float*)d_in[2];
  const float* wk = (const float*)d_in[3];
  const float* bk = (const float*)d_in[4];
  const float* wv = (const float*)d_in[5];
  const float* bv = (const float*)d_in[6];
  const float* wo = (const float*)d_in[7];
  const float* bo = (const float*)d_in[8];
  float* out = (float*)d_out;

  char* ws = (char*)d_ws;
  size_t off = 0;
  _Float16* xh  = (_Float16*)(ws + off); off += (size_t)N_TOK * C_IN * 2;    // 2 MB
  _Float16* wqT = (_Float16*)(ws + off); off += (size_t)C_IN * C_IN * 2;     // 128 KB
  _Float16* wkT = (_Float16*)(ws + off); off += (size_t)C_IN * C_IN * 2;
  _Float16* wvT = (_Float16*)(ws + off); off += (size_t)C_IN * C_IN * 2;
  _Float16* woT = (_Float16*)(ws + off); off += (size_t)C_IN * C_IN * 2;
  _Float16* qh  = (_Float16*)(ws + off); off += (size_t)NHEAD * N_TOK * DHEAD * 2; // 2 MB
  _Float16* kh  = (_Float16*)(ws + off); off += (size_t)NHEAD * N_TOK * DHEAD * 2;
  _Float16* vT  = (_Float16*)(ws + off); off += (size_t)NHEAD * DHEAD * N_TOK * 2;
  _Float16* ao  = (_Float16*)(ws + off); off += (size_t)N_TOK * C_IN * 2;    // 2 MB

  cvt_x_f16<<<(N_TOK * C_IN + 255) / 256, 256, 0, stream>>>(x, xh, N_TOK * C_IN);
  cvt_w_transpose<<<C_IN, C_IN, 0, stream>>>(wq, wqT);
  cvt_w_transpose<<<C_IN, C_IN, 0, stream>>>(wk, wkT);
  cvt_w_transpose<<<C_IN, C_IN, 0, stream>>>(wv, wvT);
  cvt_w_transpose<<<C_IN, C_IN, 0, stream>>>(wo, woT);

  qkv_proj<<<dim3(N_TOK / 16, C_IN / 16), 32, 0, stream>>>(
      xh, wqT, wkT, wvT, bq, bk, bv, qh, kh, vT);

  attn_fwd<<<dim3(N_TOK / 16, NHEAD), 32, 0, stream>>>(qh, kh, vT, ao);

  out_proj<<<dim3(N_TOK / 16, C_IN / 16), 32, 0, stream>>>(ao, woT, bo, out);
}
